// GraphTransformer_60421599920559
// MI455X (gfx1250) — matrix operand order repaired and checked by reading, not compiled, not run on hardware
//
#include <hip/hip_runtime.h>
#include <hip/hip_bf16.h>
#include <math.h>

// ---------------------------------------------------------------------------
// GraphTransformer (2x TransformerConv, heads=1) for MI455X / gfx1250.
//   GEMMs      : v_wmma_f32_16x16x32_f16, wave computes 16x16 tile, K=128.
//   Edge phase : wave32-per-edge gathers (float4 per lane), shfl reductions,
//                f32 atomics for segment softmax + scatter aggregation.
// ---------------------------------------------------------------------------

typedef __attribute__((ext_vector_type(16))) _Float16 v16h;
typedef __attribute__((ext_vector_type(8)))  _Float16 v8h;
typedef __attribute__((ext_vector_type(8)))  float    v8f;

#define NODES 50000
#define EDGES 800000
#define INC   128
#define HIDC  128
#define OUTC  40

// ----- monotone float <-> uint mapping for atomicMax on floats --------------
__device__ __forceinline__ unsigned fmap(float f) {
  unsigned u = __float_as_uint(f);
  return (u & 0x80000000u) ? ~u : (u | 0x80000000u);
}
__device__ __forceinline__ float funmap(unsigned m) {
  return __uint_as_float((m & 0x80000000u) ? (m ^ 0x80000000u) : ~m);
}
#define FMAP_NEG_INF 0x007FFFFFu   // fmap(-inf)

// ---------------------------------------------------------------------------
// Elementwise f32 -> f16 conversion.
__global__ void k_to_f16(const float* __restrict__ x, _Float16* __restrict__ xh, int n) {
  int i = blockIdx.x * blockDim.x + threadIdx.x;
  if (i < n) xh[i] = (_Float16)x[i];
}

// ---------------------------------------------------------------------------
// Pack weight W[K=128, Ncsrc] (row major) into WMMA B-fragment layout (f16).
// Fragment f = kt*NT_total + (ct_base + cm). Within a fragment, lane l holds
// 16 halves: element e = W[kt*32 + (e>>3)*16 + (l>>4)*8 + (e&7)][cm*16 + (l&15)].
__global__ void k_pack_w(const float* __restrict__ W, _Float16* __restrict__ wp,
                         int NT_total, int ct_base, int NT_mat, int Ncsrc) {
  int idx = blockIdx.x * blockDim.x + threadIdx.x;
  int total = 4 * NT_mat * 32 * 16;          // 4 k-tiles (K=128)
  if (idx >= total) return;
  int e   = idx & 15;
  int l   = (idx >> 4) & 31;
  int rem = idx >> 9;
  int cm  = rem % NT_mat;
  int kt  = rem / NT_mat;
  int n   = cm * 16 + (l & 15);
  int kk  = kt * 32 + (e >> 3) * 16 + (l >> 4) * 8 + (e & 7);
  float v = (n < Ncsrc) ? W[(size_t)kk * Ncsrc + n] : 0.0f;
  size_t o = (((size_t)kt * NT_total + (ct_base + cm)) * 32 + l) * 16 + e;
  wp[o] = (_Float16)v;
}

// Concatenate 4 bias vectors (length nsrc each) into 4 slots of slotN, 0-pad.
__global__ void k_pack_bias(float* __restrict__ dst,
                            const float* __restrict__ b0, const float* __restrict__ b1,
                            const float* __restrict__ b2, const float* __restrict__ b3,
                            int slotN, int nsrc) {
  int idx = blockIdx.x * blockDim.x + threadIdx.x;
  if (idx >= 4 * slotN) return;
  int slot = idx / slotN, c = idx % slotN;
  const float* s = (slot == 0) ? b0 : (slot == 1) ? b1 : (slot == 2) ? b2 : b3;
  dst[idx] = (c < nsrc) ? s[c] : 0.0f;
}

// ---------------------------------------------------------------------------
// WMMA GEMM:  Y[M, NT*16] = Ah[M,128] * Wpacked + bias.   One wave = one
// 16-row tile, looping over all NT column tiles; K=128 -> 4 wmma per tile.
__global__ void k_gemm_wmma(const _Float16* __restrict__ Ah,
                            const _Float16* __restrict__ Wp,
                            const float* __restrict__ bias,
                            float* __restrict__ Y,
                            int M, int NT, int ldY) {
  int wave = (blockIdx.x * blockDim.x + threadIdx.x) >> 5;
  int lane = threadIdx.x & 31;
  int row0 = wave * 16;
  if (row0 >= M) return;
  int m = lane & 15;
  int h = lane >> 4;

  // A fragments for K = 0..127 (ISA 16-bit A layout: lanes 0-15 -> K lo half,
  // lanes 16-31 -> K hi half; elems 0..7 = first 8, elems 8..15 = +16).
  v16h afr[4];
#pragma unroll
  for (int kt = 0; kt < 4; ++kt) {
    const _Float16* ap = Ah + (size_t)(row0 + m) * 128 + kt * 32 + h * 8;
    v8h lo = *(const v8h*)ap;
    v8h hi = *(const v8h*)(ap + 16);
    v16h a;
#pragma unroll
    for (int e = 0; e < 8; ++e) { a[e] = lo[e]; a[e + 8] = hi[e]; }
    afr[kt] = a;
  }

  int n = lane & 15;
  for (int ct = 0; ct < NT; ++ct) {
    v8f c = {};
#pragma unroll
    for (int kt = 0; kt < 4; ++kt) {
      v16h b = *(const v16h*)(Wp + (((size_t)kt * NT + ct) * 32 + lane) * 16);
      c = __builtin_amdgcn_wmma_f32_16x16x32_f16(false, afr[kt], false, b,
                                                 (short)0, c, false, false);
    }
    int col0 = ct * 16;
    float bb = bias[col0 + n];
#pragma unroll
    for (int r = 0; r < 8; ++r) {
      Y[(size_t)(row0 + h * 8 + r) * ldY + col0 + n] = c[r] + bb;
    }
  }
}

// ---------------------------------------------------------------------------
__global__ void k_init_seg(unsigned* __restrict__ mseg, float* __restrict__ sseg, int n) {
  int i = blockIdx.x * blockDim.x + threadIdx.x;
  if (i < n) { mseg[i] = FMAP_NEG_INF; sseg[i] = 0.0f; }
}

// scores[e] = dot(q[dst], k[src]) * scale ; segment max into mseg (monotone u32).
// One wave per edge; lanes < dvec do float4 partial dots.
__global__ void k_edge_scores(const float* __restrict__ qk, int qoff, int koff,
                              int ld, int dvec, float scale,
                              const int* __restrict__ src, const int* __restrict__ dst,
                              float* __restrict__ scores, unsigned* __restrict__ mseg,
                              int E) {
  int gt = blockIdx.x * blockDim.x + threadIdx.x;
  int e = gt >> 5, lane = gt & 31;
  if (e >= E) return;
  int s = src[e], d = dst[e];
  float acc = 0.0f;
  if (lane < dvec) {
    float4 qv = *(const float4*)(qk + (size_t)d * ld + qoff + lane * 4);
    float4 kv = *(const float4*)(qk + (size_t)s * ld + koff + lane * 4);
    acc = qv.x * kv.x + qv.y * kv.y + qv.z * kv.z + qv.w * kv.w;
  }
#pragma unroll
  for (int off = 16; off; off >>= 1) acc += __shfl_xor(acc, off, 32);
  float sc = acc * scale;
  if (lane == 0) {
    scores[e] = sc;
    atomicMax(mseg + d, fmap(sc));
  }
}

// eval[e] = exp(score - m[dst]); segment sum into sseg.
__global__ void k_edge_exp(const float* __restrict__ scores,
                           const unsigned* __restrict__ mseg,
                           const int* __restrict__ dst,
                           float* __restrict__ eval_, float* __restrict__ sseg, int E) {
  int e = blockIdx.x * blockDim.x + threadIdx.x;
  if (e >= E) return;
  int d = dst[e];
  float ev = __expf(scores[e] - funmap(mseg[d]));
  eval_[e] = ev;
  atomicAdd(sseg + d, ev);
}

// agg[dst,:] += (eval/s[dst]) * v[src,:]   (agg slot pre-loaded with skip term).
__global__ void k_edge_agg(const float* __restrict__ eval_, const float* __restrict__ sseg,
                           float* __restrict__ buf, int voff, int aoff, int ld, int dvec,
                           const int* __restrict__ src, const int* __restrict__ dst, int E) {
  int gt = blockIdx.x * blockDim.x + threadIdx.x;
  int e = gt >> 5, lane = gt & 31;
  if (e >= E) return;
  int s = src[e], d = dst[e];
  float alpha = eval_[e] / sseg[d];
  if (lane < dvec) {
    float4 vv = *(const float4*)(buf + (size_t)s * ld + voff + lane * 4);
    float* ap = buf + (size_t)d * ld + aoff + lane * 4;
    atomicAdd(ap + 0, alpha * vv.x);
    atomicAdd(ap + 1, alpha * vv.y);
    atomicAdd(ap + 2, alpha * vv.z);
    atomicAdd(ap + 3, alpha * vv.w);
  }
}

// x2h = f16(relu(layer-1 output slot))
__global__ void k_relu_f16(const float* __restrict__ buf, int aoff, int ld,
                           _Float16* __restrict__ xh, int n) {
  int idx = blockIdx.x * blockDim.x + threadIdx.x;
  if (idx >= n) return;
  int i = idx >> 7, c = idx & 127;
  float v = buf[(size_t)i * ld + aoff + c];
  xh[idx] = (_Float16)fmaxf(v, 0.0f);
}

// out[i,:] = log_softmax(relu(y[i,:])) over 40 cols; one wave per node.
__global__ void k_final(const float* __restrict__ buf, int aoff, int ld,
                        float* __restrict__ out, int N) {
  int gt = blockIdx.x * blockDim.x + threadIdx.x;
  int i = gt >> 5, lane = gt & 31;
  if (i >= N) return;
  const float* row = buf + (size_t)i * ld + aoff;
  float a0 = fmaxf(row[lane], 0.0f);                                  // lane < 32 < 40
  float a1 = (lane < 8) ? fmaxf(row[lane + 32], 0.0f) : -INFINITY;    // cols 32..39
  float mx = fmaxf(a0, a1);
#pragma unroll
  for (int off = 16; off; off >>= 1) mx = fmaxf(mx, __shfl_xor(mx, off, 32));
  float sm = __expf(a0 - mx) + ((lane < 8) ? __expf(a1 - mx) : 0.0f);
#pragma unroll
  for (int off = 16; off; off >>= 1) sm += __shfl_xor(sm, off, 32);
  float ls = __logf(sm);
  out[(size_t)i * 40 + lane] = a0 - mx - ls;
  if (lane < 8) out[(size_t)i * 40 + lane + 32] = a1 - mx - ls;
}

// ---------------------------------------------------------------------------
extern "C" void kernel_launch(void* const* d_in, const int* in_sizes, int n_in,
                              void* d_out, int out_size, void* d_ws, size_t ws_size,
                              hipStream_t stream) {
  (void)in_sizes; (void)n_in; (void)out_size; (void)ws_size;
  const float* feat = (const float*)d_in[0];
  const int*   eidx = (const int*)d_in[1];
  const int* src = eidx;
  const int* dst = eidx + EDGES;
  const float* Wq1 = (const float*)d_in[2];  const float* bq1 = (const float*)d_in[3];
  const float* Wk1 = (const float*)d_in[4];  const float* bk1 = (const float*)d_in[5];
  const float* Wv1 = (const float*)d_in[6];  const float* bv1 = (const float*)d_in[7];
  const float* Ws1 = (const float*)d_in[8];  const float* bs1 = (const float*)d_in[9];
  const float* Wq2 = (const float*)d_in[10]; const float* bq2 = (const float*)d_in[11];
  const float* Wk2 = (const float*)d_in[12]; const float* bk2 = (const float*)d_in[13];
  const float* Wv2 = (const float*)d_in[14]; const float* bv2 = (const float*)d_in[15];
  const float* Ws2 = (const float*)d_in[16]; const float* bs2 = (const float*)d_in[17];
  float* out = (float*)d_out;

  // ---- workspace layout (bump allocator, 256B aligned) ----
  char* ws = (char*)d_ws;
  size_t off = 0;
  auto take = [&](size_t bytes) -> char* {
    char* p = ws + off;
    off = (off + bytes + 255) & ~(size_t)255;
    return p;
  };
  _Float16* xh    = (_Float16*)take((size_t)NODES * 128 * 2);   // f16 features
  _Float16* w1p   = (_Float16*)take((size_t)4 * 32 * 512 * 2);  // packed layer1 W (NT=32)
  float*    b1    = (float*)   take(512 * 4);
  _Float16* w2p   = (_Float16*)take((size_t)4 * 12 * 512 * 2);  // packed layer2 W (NT=12)
  float*    b2    = (float*)   take(192 * 4);
  float*    scores= (float*)   take((size_t)EDGES * 4);
  float*    eval_ = (float*)   take((size_t)EDGES * 4);
  unsigned* mseg  = (unsigned*)take((size_t)NODES * 4);
  float*    sseg  = (float*)   take((size_t)NODES * 4);
  _Float16* x2h   = (_Float16*)take((size_t)NODES * 128 * 2);   // f16 layer-1 output
  float*    qkvs1 = (float*)   take((size_t)NODES * 512 * 4);   // q|k|v|agg  (ld=512)
  float*    qkvs2 = qkvs1;                                      // alias: ld=192, layer1 dead

  const int T256 = 256;
  const float scale1 = 0.08838834764831845f;   // 1/sqrt(128)
  const float scale2 = 0.15811388300841897f;   // 1/sqrt(40)

  // ---- prep: f16 features, packed weights, biases ----
  k_to_f16<<<(NODES * 128 + T256 - 1) / T256, T256, 0, stream>>>(feat, xh, NODES * 128);

  int p1 = (4 * 8 * 512 + T256 - 1) / T256;    // per layer-1 matrix
  k_pack_w<<<p1, T256, 0, stream>>>(Wq1, w1p, 32, 0,  8, 128);
  k_pack_w<<<p1, T256, 0, stream>>>(Wk1, w1p, 32, 8,  8, 128);
  k_pack_w<<<p1, T256, 0, stream>>>(Wv1, w1p, 32, 16, 8, 128);
  k_pack_w<<<p1, T256, 0, stream>>>(Ws1, w1p, 32, 24, 8, 128);
  int p2 = (4 * 3 * 512 + T256 - 1) / T256;    // per layer-2 matrix (Nc=40 pad 48)
  k_pack_w<<<p2, T256, 0, stream>>>(Wq2, w2p, 12, 0, 3, 40);
  k_pack_w<<<p2, T256, 0, stream>>>(Wk2, w2p, 12, 3, 3, 40);
  k_pack_w<<<p2, T256, 0, stream>>>(Wv2, w2p, 12, 6, 3, 40);
  k_pack_w<<<p2, T256, 0, stream>>>(Ws2, w2p, 12, 9, 3, 40);
  k_pack_bias<<<(512 + T256 - 1) / T256, T256, 0, stream>>>(b1, bq1, bk1, bv1, bs1, 128, 128);
  k_pack_bias<<<(192 + T256 - 1) / T256, T256, 0, stream>>>(b2, bq2, bk2, bv2, bs2, 48, 40);

  // ---- layer 1 ----
  int rowTiles = NODES / 16;                                   // 3125 (exact)
  int gemmBlocks = (rowTiles + 7) / 8;                         // 8 waves / block
  k_gemm_wmma<<<gemmBlocks, T256, 0, stream>>>(xh, w1p, b1, qkvs1, NODES, 32, 512);

  int edgeBlocks = (EDGES * 32 + T256 - 1) / T256;
  k_init_seg<<<(NODES + T256 - 1) / T256, T256, 0, stream>>>(mseg, sseg, NODES);
  k_edge_scores<<<edgeBlocks, T256, 0, stream>>>(qkvs1, 0, 128, 512, 32, scale1,
                                                 src, dst, scores, mseg, EDGES);
  k_edge_exp<<<(EDGES + T256 - 1) / T256, T256, 0, stream>>>(scores, mseg, dst,
                                                             eval_, sseg, EDGES);
  k_edge_agg<<<edgeBlocks, T256, 0, stream>>>(eval_, sseg, qkvs1, 256, 384, 512, 32,
                                              src, dst, EDGES);
  k_relu_f16<<<(NODES * 128 + T256 - 1) / T256, T256, 0, stream>>>(qkvs1, 384, 512,
                                                                   x2h, NODES * 128);

  // ---- layer 2 (slots of 48 cols: q@0, k@48, v@96, agg@144; valid 40) ----
  k_gemm_wmma<<<gemmBlocks, T256, 0, stream>>>(x2h, w2p, b2, qkvs2, NODES, 12, 192);
  k_init_seg<<<(NODES + T256 - 1) / T256, T256, 0, stream>>>(mseg, sseg, NODES);
  k_edge_scores<<<edgeBlocks, T256, 0, stream>>>(qkvs2, 0, 48, 192, 10, scale2,
                                                 src, dst, scores, mseg, EDGES);
  k_edge_exp<<<(EDGES + T256 - 1) / T256, T256, 0, stream>>>(scores, mseg, dst,
                                                             eval_, sseg, EDGES);
  k_edge_agg<<<edgeBlocks, T256, 0, stream>>>(eval_, sseg, qkvs2, 96, 144, 192, 10,
                                              src, dst, EDGES);

  // ---- relu + log_softmax -> d_out ----
  k_final<<<(NODES * 32 + T256 - 1) / T256, T256, 0, stream>>>(qkvs2, 144, 192, out, NODES);
}
